// QuantumLayer_86569360818672
// MI455X (gfx1250) — matrix operand order repaired and checked by reading, not compiled, not run on hardware
//
#include <hip/hip_runtime.h>
#include <hip/hip_bf16.h>

#define NQ      12
#define DIM     4096      // 2^12
#define BATCH   16384
#define NLAYERS 6

typedef __attribute__((ext_vector_type(16))) __bf16 v16bf;
typedef __attribute__((ext_vector_type(8)))  __bf16 v8bf;
typedef __attribute__((ext_vector_type(8)))  float  v8f;

// Async 16-byte global -> LDS copy (CDNA5, tracked by ASYNCcnt).
__device__ __forceinline__ void async16(unsigned ldsAddr, const void* g) {
    asm volatile("global_load_async_to_lds_b128 %0, %1, off"
                 :: "v"(ldsAddr), "v"(g) : "memory");
}

// ---------------------------------------------------------------------------
// Kernel 0: zero the output (harness poisons d_out with 0xAA).
// ---------------------------------------------------------------------------
__global__ void qnn_zero(float* __restrict__ out, int n) {
    int i = blockIdx.x * 256 + threadIdx.x;
    if (i < n) out[i] = 0.0f;
}

// ---------------------------------------------------------------------------
// Kernel 1: UT[k][m] = (weight-layers circuit applied to basis e_k)[m], bf16.
// One block per basis state; 16KB state in LDS. Qubit i = bit (11-i).
// ---------------------------------------------------------------------------
__global__ __launch_bounds__(256) void qnn_build_u(const float* __restrict__ w,
                                                   __bf16* __restrict__ UT) {
    __shared__ float st[DIM];
    const int k   = blockIdx.x;
    const int tid = threadIdx.x;

    for (int m = tid; m < DIM; m += 256) st[m] = (m == k) ? 1.0f : 0.0f;
    __syncthreads();

    for (int l = 0; l < NLAYERS; ++l) {
        for (int i = 0; i < NQ; ++i) {            // CNOT ring (sequential)
            const unsigned cb = 1u << (11 - i);
            const unsigned tb = 1u << (11 - ((i + 1) % NQ));
            const unsigned hi = cb > tb ? cb : tb;
            const unsigned lo = cb > tb ? tb : cb;
            for (int p = tid; p < DIM / 4; p += 256) {
                unsigned m = (unsigned)p;
                m = ((m & ~(lo - 1)) << 1) | (m & (lo - 1));
                m = ((m & ~(hi - 1)) << 1) | (m & (hi - 1));
                m |= cb;
                float a = st[m], b = st[m | tb];
                st[m] = b; st[m | tb] = a;
            }
            __syncthreads();
        }
        for (int i = 0; i < NQ; ++i) {            // shared-weight RY layer
            const float th = 0.5f * w[l * NQ + i];
            const float c = cosf(th), s = sinf(th);
            const unsigned b = 1u << (11 - i);
            for (int p = tid; p < DIM / 2; p += 256) {
                unsigned m0 = (((unsigned)p & ~(b - 1)) << 1) | ((unsigned)p & (b - 1));
                unsigned m1 = m0 | b;
                float a0 = st[m0], a1 = st[m1];
                st[m0] = c * a0 - s * a1;
                st[m1] = s * a0 + c * a1;
            }
            __syncthreads();
        }
    }
    for (int m = tid; m < DIM; m += 256)
        UT[(size_t)k * DIM + m] = (__bf16)st[m];
}

// ---------------------------------------------------------------------------
// Kernel 2: Psi[b][k] = Hi[b][k>>6] * Lo[b][k&63] (encoding product state).
// ---------------------------------------------------------------------------
__global__ __launch_bounds__(128) void qnn_build_psi(const float* __restrict__ x,
                                                     __bf16* __restrict__ Psi) {
    __shared__ float c[NQ], s[NQ], hi[64], lo[64];
    const int b   = blockIdx.x;
    const int tid = threadIdx.x;
    if (tid < NQ) {
        float th = 0.5f * x[b * NQ + tid];
        c[tid] = cosf(th); s[tid] = sinf(th);
    }
    __syncthreads();
    if (tid < 64) {
        float p = 1.0f;
        #pragma unroll
        for (int i = 0; i < 6; ++i) p *= ((tid >> (5 - i)) & 1) ? s[i] : c[i];
        hi[tid] = p;
    } else {
        int t = tid - 64;
        float p = 1.0f;
        #pragma unroll
        for (int i = 0; i < 6; ++i) p *= ((t >> (5 - i)) & 1) ? s[6 + i] : c[6 + i];
        lo[t] = p;
    }
    __syncthreads();
    __bf16* row = Psi + (size_t)b * DIM;
    for (int m = tid; m < DIM; m += 128)
        row[m] = (__bf16)(hi[m >> 6] * lo[m & 63]);
}

// ---------------------------------------------------------------------------
// Kernel 3: Final = Psi x UT via WMMA bf16 with async-LDS double buffering,
// fused signed reduction  out[b][i] += sum_n Final[b][n]^2 * (1-2*bit_i(n)).
// Block 256 thr (8 waves, 4M x 2N), tile 128x128, K-chunk 64 (2 WMMA steps).
// ---------------------------------------------------------------------------
__global__ __launch_bounds__(256) void qnn_gemm_expect(const __bf16* __restrict__ Psi,
                                                       const __bf16* __restrict__ UT,
                                                       float* __restrict__ out) {
    __shared__ char smem[65536];   // [A0 16K][A1 16K][B0 16K][B1 16K]; reused as epilogue tile
    const int tid   = threadIdx.x;
    const int lane  = tid & 31;
    const int wave  = tid >> 5;
    const int waveM = wave >> 1;   // 0..3 -> 32-row strip
    const int waveN = wave & 1;    // 0..1 -> 64-col strip
    const int blockRow0 = blockIdx.x * 128;
    const int nBase     = blockIdx.y * 128;

    const unsigned ldsBase = (unsigned)(size_t)(&smem[0]); // low 32 bits = LDS offset
    const unsigned Aoff[2] = { 0u, 16384u };
    const unsigned Boff[2] = { 32768u, 49152u };

    const v8f vz = {0.f,0.f,0.f,0.f,0.f,0.f,0.f,0.f};
    v8f acc[8];                      // [sub(2)][t(4)]
    #pragma unroll
    for (int j = 0; j < 8; ++j) acc[j] = vz;

    // stage one 64-wide K chunk: A 128x64 bf16 (16KB rows of 128B),
    // B 64x128 bf16 (16KB rows of 256B); 8 async b128 per thread.
    auto stage = [&](int k0, int buf) {
        #pragma unroll
        for (int j = 0; j < 4; ++j) {            // A: 1024 x 16B chunks
            int id  = tid + j * 256;
            int row = id >> 3, c = id & 7;
            const __bf16* g = Psi + (size_t)(blockRow0 + row) * DIM + k0 + c * 8;
            unsigned l = ldsBase + Aoff[buf] + (unsigned)(row * 128)
                       + (unsigned)(((c ^ ((row >> 1) & 7)) & 7) * 16);
            async16(l, g);
        }
        #pragma unroll
        for (int j = 0; j < 4; ++j) {            // B: 1024 x 16B chunks
            int id = tid + j * 256;
            int k  = id >> 4, c = id & 15;
            const __bf16* g = UT + (size_t)(k0 + k) * DIM + nBase + c * 8;
            unsigned sw = (unsigned)(((c ^ (k & 15) ^ (((k >> 4) & 1) << 3)) & 15) * 16);
            unsigned l = ldsBase + Boff[buf] + (unsigned)(k * 256) + sw;
            async16(l, g);
        }
    };

    stage(0, 0);
    #pragma unroll 1
    for (int kt = 0; kt < 64; ++kt) {
        const int buf = kt & 1;
        if (kt < 63) {
            stage((kt + 1) * 64, buf ^ 1);
            asm volatile("s_wait_asynccnt 0x8" ::: "memory");   // chunk kt landed
        } else {
            asm volatile("s_wait_asynccnt 0x0" ::: "memory");
        }
        __syncthreads();

        const char* Ab = smem + Aoff[buf];
        const char* Bb = smem + Boff[buf];
        #pragma unroll
        for (int s = 0; s < 2; ++s) {            // two K=32 substeps
            v16bf afr[2];
            #pragma unroll
            for (int sub = 0; sub < 2; ++sub) {  // two 16-row A frags
                int row = waveM * 32 + sub * 16 + (lane & 15);
                int c0  = s * 4 + (lane >> 4);
                int g   = (row >> 1) & 7;
                v8bf a0 = *(const v8bf*)(Ab + row * 128 + ((c0 ^ g) & 7) * 16);
                v8bf a1 = *(const v8bf*)(Ab + row * 128 + (((c0 + 2) ^ g) & 7) * 16);
                afr[sub] = __builtin_shufflevector(a0, a1,
                            0,1,2,3,4,5,6,7,8,9,10,11,12,13,14,15);
            }
            const int hb  = lane >> 4;
            v16bf bfr[4];
            #pragma unroll
            for (int t = 0; t < 4; ++t) {        // four 16-col B frags
                int n = waveN * 64 + t * 16 + (lane & 15);
                #pragma unroll
                for (int e = 0; e < 16; ++e) {
                    int kk  = s * 32 + (e < 8 ? 8 * hb + e : 16 + 8 * hb + (e - 8));
                    int ksw = (kk & 15) ^ (((kk >> 4) & 1) << 3);
                    bfr[t][e] = *(const __bf16*)(Bb + kk * 256 + (((n >> 3) ^ ksw) & 15) * 16 + (n & 7) * 2);
                }
            }
            #pragma unroll
            for (int sub = 0; sub < 2; ++sub)
                #pragma unroll
                for (int t = 0; t < 4; ++t)
                    acc[sub * 4 + t] = __builtin_amdgcn_wmma_f32_16x16x32_bf16(
                        false, afr[sub], false, bfr[t], (short)0,
                        acc[sub * 4 + t], false, false);
        }
        __syncthreads();
    }

    // Epilogue: two column halves through a padded 128x65 f32 LDS tile,
    // partial sums kept in registers, single atomicAdd per (row,i).
    float* tile = (float*)smem;
    float psum[6];
    #pragma unroll
    for (int j = 0; j < 6; ++j) psum[j] = 0.0f;

    #pragma unroll 1
    for (int h = 0; h < 2; ++h) {
        __syncthreads();
        if (waveN == h) {
            const int colb = lane & 15;
            const int rb   = waveM * 32 + ((lane >> 4) << 3);
            #pragma unroll
            for (int sub = 0; sub < 2; ++sub)
                #pragma unroll
                for (int t = 0; t < 4; ++t)
                    #pragma unroll
                    for (int r = 0; r < 8; ++r)
                        tile[(rb + sub * 16 + r) * 65 + colb + t * 16] = acc[sub * 4 + t][r];
        }
        __syncthreads();
        for (int j = 0; j < 6; ++j) {
            int pr = tid + j * 256;              // 128 rows x 12 qubits
            int row = pr / NQ, i = pr % NQ, sh = 11 - i;
            float sum = 0.0f;
            #pragma unroll 8
            for (int n = 0; n < 64; ++n) {
                float v = tile[row * 65 + n];
                sum += ((((nBase + h * 64 + n) >> sh) & 1) ? -v : v) * v;
            }
            psum[j] += sum;
        }
    }
    #pragma unroll
    for (int j = 0; j < 6; ++j) {
        int pr = tid + j * 256;
        int row = pr / NQ, i = pr % NQ;
        atomicAdd(&out[(size_t)(blockRow0 + row) * NQ + i], psum[j]);
    }
}

// ---------------------------------------------------------------------------
extern "C" void kernel_launch(void* const* d_in, const int* in_sizes, int n_in,
                              void* d_out, int out_size, void* d_ws, size_t ws_size,
                              hipStream_t stream) {
    const float* x = (const float*)d_in[0];   // [16384][12]
    const float* w = (const float*)d_in[1];   // [6][12]
    float* out = (float*)d_out;               // [16384][12]

    __bf16* UT  = (__bf16*)d_ws;                                   // 32 MB
    __bf16* Psi = (__bf16*)((char*)d_ws + (size_t)DIM * DIM * 2);  // 128 MB

    qnn_zero<<<(BATCH * NQ + 255) / 256, 256, 0, stream>>>(out, BATCH * NQ);
    qnn_build_u<<<DIM, 256, 0, stream>>>(w, UT);
    qnn_build_psi<<<BATCH, 128, 0, stream>>>(x, Psi);

    dim3 grid(BATCH / 128, DIM / 128);   // 128 x 32 blocks
    qnn_gemm_expect<<<grid, 256, 0, stream>>>(Psi, UT, out);
}
